// ModalitySpecificAttention_59622736003255
// MI455X (gfx1250) — matrix operand order repaired and checked by reading, not compiled, not run on hardware
//
#include <hip/hip_runtime.h>
#include <hip/hip_bf16.h>

// ---------------- problem constants ----------------
#define BATCH   2
#define SEQ     2048
#define DM      1024
#define NHEAD   16
#define HDIM    64
#define ROWS    (BATCH * SEQ)        // 4096 token rows
#define LN_EPS  1e-5f
#define QSCALE  0.125f               // HDIM^-0.5

typedef _Float16 h4v  __attribute__((ext_vector_type(4)));
typedef _Float16 h8v  __attribute__((ext_vector_type(8)));
typedef _Float16 v16h __attribute__((ext_vector_type(16)));
typedef float    v8f  __attribute__((ext_vector_type(8)));

// ---------------- WMMA helpers (CDNA5, wave32) ----------------
// A tile 16x32 f16 (M x K).  ISA layout: lanes 0-15 hold row M=lane with
// K = 0..7 and 16..23; lanes 16-31 hold the same row with K = 8..15 and
// 24..31.  => two contiguous 16B loads per lane.
__device__ __forceinline__ v16h loadA16x32(const _Float16* __restrict__ p, int ld, int lane) {
  int row = lane & 15;
  int cb  = (lane < 16) ? 0 : 8;
  const _Float16* q = p + (size_t)row * ld + cb;
  h8v lo = *(const h8v*)(q);
  h8v hi = *(const h8v*)(q + 16);
  v16h a;
#pragma unroll
  for (int i = 0; i < 8; ++i) { a[i] = lo[i]; a[i + 8] = hi[i]; }
  return a;
}

// B tile 32x16 f16 (K x N): column n of B is a contiguous K-run at cp + n*ld.
// Lanes 0-15: column n, K=0..7,16..23; lanes 16-31: column n, K=8..15,24..31.
__device__ __forceinline__ v16h loadB32x16(const _Float16* __restrict__ cp, int ld, int lane) {
  int col = lane & 15;
  int kb  = (lane < 16) ? 0 : 8;
  const _Float16* q = cp + (size_t)col * ld + kb;
  h8v lo = *(const h8v*)(q);
  h8v hi = *(const h8v*)(q + 16);
  v16h b;
#pragma unroll
  for (int i = 0; i < 8; ++i) { b[i] = lo[i]; b[i + 8] = hi[i]; }
  return b;
}

__device__ __forceinline__ v8f wmma_f16(v16h a, v16h b, v8f c) {
  return __builtin_amdgcn_wmma_f32_16x16x32_f16(false, a, false, b, (short)0, c, false, false);
}

// async 16B global -> LDS copy (gfx1250), tracked by ASYNCcnt
__device__ __forceinline__ void async_copy16(const _Float16* g, _Float16* l) {
  unsigned lds = (unsigned)(unsigned long long)l;          // low 32 bits = LDS byte offset
  unsigned long long ga = (unsigned long long)g;
  asm volatile("global_load_async_to_lds_b128 %0, %1, off"
               :: "v"(lds), "v"(ga) : "memory");
}

// ---------------- LayerNorm: f32 row -> f16 row ----------------
__global__ __launch_bounds__(256) void ln_kernel(const float* __restrict__ x,
                                                 const float* __restrict__ g,
                                                 const float* __restrict__ bb,
                                                 _Float16* __restrict__ o) {
  int row  = blockIdx.x;
  int tid  = threadIdx.x;
  int lane = tid & 31;
  int wv   = tid >> 5;
  const float4* xr = (const float4*)(x + (size_t)row * DM);
  float4 v = xr[tid];
  float s = v.x + v.y + v.z + v.w;
  float q = v.x * v.x + v.y * v.y + v.z * v.z + v.w * v.w;
#pragma unroll
  for (int mk = 1; mk < 32; mk <<= 1) {
    s += __shfl_xor(s, mk, 32);
    q += __shfl_xor(q, mk, 32);
  }
  __shared__ float ss[8], sq[8];
  if (lane == 0) { ss[wv] = s; sq[wv] = q; }
  __syncthreads();
  float ts = 0.f, tq = 0.f;
#pragma unroll
  for (int i = 0; i < 8; ++i) { ts += ss[i]; tq += sq[i]; }
  float mu   = ts * (1.f / DM);
  float var  = tq * (1.f / DM) - mu * mu;
  float rstd = rsqrtf(var + LN_EPS);
  float4 gv = ((const float4*)g)[tid];
  float4 bv = ((const float4*)bb)[tid];
  h4v out;
  out[0] = (_Float16)((v.x - mu) * rstd * gv.x + bv.x);
  out[1] = (_Float16)((v.y - mu) * rstd * gv.y + bv.y);
  out[2] = (_Float16)((v.z - mu) * rstd * gv.z + bv.z);
  out[3] = (_Float16)((v.w - mu) * rstd * gv.w + bv.w);
  ((h4v*)(o + (size_t)row * DM))[tid] = out;
}

// ---------------- f32 -> f16 converter (weights) ----------------
__global__ __launch_bounds__(256) void cvt_f32_f16(const float* __restrict__ in,
                                                   _Float16* __restrict__ out, int nv4) {
  int i = blockIdx.x * 256 + threadIdx.x;
  if (i >= nv4) return;
  float4 v = ((const float4*)in)[i];
  h4v o;
  o[0] = (_Float16)v.x; o[1] = (_Float16)v.y;
  o[2] = (_Float16)v.z; o[3] = (_Float16)v.w;
  ((h4v*)out)[i] = o;
}

// ---------------- NT GEMM: out[r,c] = sum_d A[r,d] * W[c,d] + bias[c] ------
// 2x ping-pong software pipeline (WMMA groups consume tiles loaded one group
// earlier; residual v_dual_mov copies co-execute with the XDL pipe).
__global__ __launch_bounds__(256) void gemm_nt(const _Float16* __restrict__ A,
                                               const _Float16* __restrict__ W,
                                               const float* __restrict__ bias,
                                               void* __restrict__ Out,
                                               float scale, int mode) {
  const int K = DM;
  int lane = threadIdx.x & 31;
  int w    = blockIdx.x * 8 + (threadIdx.x >> 5);
  int mrow = w >> 4;
  int ncol = w & 15;
  const _Float16* Ap    = A + (size_t)mrow * 16 * K;
  const _Float16* Wbase = W + (size_t)(ncol * 64) * K;

  v8f acc[4] = {};
  v16h a0 = loadA16x32(Ap, K, lane);
  v16h b0[4];
#pragma unroll
  for (int dt = 0; dt < 4; ++dt)
    b0[dt] = loadB32x16(Wbase + (size_t)dt * 16 * K, K, lane);
  v16h a1 = loadA16x32(Ap + 32, K, lane);
  v16h b1[4];
#pragma unroll
  for (int dt = 0; dt < 4; ++dt)
    b1[dt] = loadB32x16(Wbase + (size_t)dt * 16 * K + 32, K, lane);

  for (int kc = 0; kc < K - 64; kc += 64) {
#pragma unroll
    for (int dt = 0; dt < 4; ++dt)
      acc[dt] = wmma_f16(a0, b0[dt], acc[dt]);
    a0 = loadA16x32(Ap + kc + 64, K, lane);
#pragma unroll
    for (int dt = 0; dt < 4; ++dt)
      b0[dt] = loadB32x16(Wbase + (size_t)dt * 16 * K + kc + 64, K, lane);
#pragma unroll
    for (int dt = 0; dt < 4; ++dt)
      acc[dt] = wmma_f16(a1, b1[dt], acc[dt]);
    a1 = loadA16x32(Ap + kc + 96, K, lane);
#pragma unroll
    for (int dt = 0; dt < 4; ++dt)
      b1[dt] = loadB32x16(Wbase + (size_t)dt * 16 * K + kc + 96, K, lane);
  }
#pragma unroll
  for (int dt = 0; dt < 4; ++dt)
    acc[dt] = wmma_f16(a0, b0[dt], acc[dt]);
#pragma unroll
  for (int dt = 0; dt < 4; ++dt)
    acc[dt] = wmma_f16(a1, b1[dt], acc[dt]);

  int n     = lane & 15;
  int half8 = (lane < 16) ? 0 : 8;
  if (mode == 0) {
    _Float16* o = (_Float16*)Out;
#pragma unroll
    for (int dt = 0; dt < 4; ++dt) {
      int col = ncol * 64 + dt * 16 + n;
      float bv = bias[col];
#pragma unroll
      for (int j = 0; j < 8; ++j) {
        int row = mrow * 16 + half8 + j;
        o[(size_t)row * DM + col] = (_Float16)((acc[dt][j] + bv) * scale);
      }
    }
  } else if (mode == 1) {
    _Float16* o = (_Float16*)Out;
#pragma unroll
    for (int dt = 0; dt < 4; ++dt) {
      int col = ncol * 64 + dt * 16 + n;
      float bv = bias[col];
      int h = col >> 6, d = col & 63;
#pragma unroll
      for (int j = 0; j < 8; ++j) {
        int row = mrow * 16 + half8 + j;
        int b = row >> 11, sIdx = row & (SEQ - 1);
        o[(((size_t)(b * NHEAD + h)) * HDIM + d) * SEQ + sIdx] =
            (_Float16)(acc[dt][j] + bv);
      }
    }
  } else {
    float* o = (float*)Out;
#pragma unroll
    for (int dt = 0; dt < 4; ++dt) {
      int col = ncol * 64 + dt * 16 + n;
      float bv = bias[col];
#pragma unroll
      for (int j = 0; j < 8; ++j) {
        int row = mrow * 16 + half8 + j;
        o[(size_t)row * DM + col] = acc[dt][j] + bv;
      }
    }
  }
}

// ---------------- fused causal attention, LDS-staged K/V ----------------
// grid = (SEQ/128, BATCH*NHEAD), block = 256 (8 waves), wave w owns 16 rows.
// All waves share (b,h): K/V tiles are staged block-cooperatively into LDS
// with async global->LDS copies (double-buffered stages of 4 s-tiles), so
// each tile is fetched once per block instead of once per wave.
__global__ __launch_bounds__(256) void attn_kernel(const _Float16* __restrict__ qb,
                                                   const _Float16* __restrict__ kb,
                                                   const _Float16* __restrict__ vt,
                                                   _Float16* __restrict__ ctx,
                                                   float* __restrict__ attn) {
  __shared__ alignas(16) _Float16 kst[2][64 * 64];   // [stage buf][s 0..63][d 0..63]
  __shared__ alignas(16) _Float16 vst[2][64 * 64];   // [stage buf][d 0..63][s 0..63]
  __shared__ alignas(16) _Float16 plds[8][16 * 40];  // per-wave P transpose tile
  int tid   = threadIdx.x;
  int lane  = tid & 31;
  int wv    = tid >> 5;
  int bh    = blockIdx.y;
  int b     = bh >> 4, h = bh & 15;
  int t0b   = blockIdx.x * 128;
  int t0    = t0b + wv * 16;
  int nt    = (t0 >> 4) + 1;           // per-wave causal tile count
  int NSTAGE = (t0b >> 6) + 2;         // block stages of 4 tiles (uniform)
  int n     = lane & 15;
  int half8 = (lane < 16) ? 0 : 8;

  const _Float16* qrow  = qb + ((size_t)(b * SEQ) + t0) * DM + h * HDIM;
  const _Float16* kbase = kb + (size_t)(b * SEQ) * DM + h * HDIM;
  const _Float16* vhead = vt + (size_t)bh * HDIM * SEQ;
  float*          arow  = attn + (size_t)bh * SEQ * SEQ;

  // staging thread roles: 64 rows x 4 segments of 16B (+4 more for row len 64)
  int srow = tid >> 2, sseg = tid & 3;

  v16h qa0 = loadA16x32(qrow, DM, lane);
  v16h qa1 = loadA16x32(qrow + 32, DM, lane);

  float m[8], r[8];
#pragma unroll
  for (int j = 0; j < 8; ++j) { m[j] = -__builtin_inff(); r[j] = 0.f; }

  // ================= sweep A: online softmax stats =================
  // prologue: stage 0 K copies (2 async instrs per thread)
  {
    const _Float16* g = kbase + (size_t)srow * DM + sseg * 8;
    async_copy16(g,      &kst[0][srow * 64 + sseg * 8]);
    async_copy16(g + 32, &kst[0][srow * 64 + sseg * 8 + 32]);
  }
  for (int sg = 0; sg < NSTAGE; ++sg) {
    int buf = sg & 1;
    if (sg + 1 < NSTAGE) {
      const _Float16* g = kbase + (size_t)((sg + 1) * 64 + srow) * DM + sseg * 8;
      async_copy16(g,      &kst[buf ^ 1][srow * 64 + sseg * 8]);
      async_copy16(g + 32, &kst[buf ^ 1][srow * 64 + sseg * 8 + 32]);
      asm volatile("s_wait_asynccnt 2" ::: "memory");
    } else {
      asm volatile("s_wait_asynccnt 0" ::: "memory");
    }
    __syncthreads();                       // stage sg visible block-wide
    int sbase = sg * 64;
    if (sbase < nt * 16) {                 // any causal work for this wave
      v8f c[4];
#pragma unroll
      for (int u = 0; u < 4; ++u) {
        const _Float16* cp = &kst[buf][(u * 16) * 64];
        v16h k0 = loadB32x16(cp, 64, lane);
        v16h k1 = loadB32x16(cp + 32, 64, lane);
        v8f t = {};
        t = wmma_f16(qa0, k0, t);
        t = wmma_f16(qa1, k1, t);
        c[u] = t;
      }
#pragma unroll
      for (int j = 0; j < 8; ++j) {
        int row = t0 + half8 + j;
        float cv[4];
#pragma unroll
        for (int u = 0; u < 4; ++u) {
          float x = c[u][j];
          if (sbase + u * 16 + n > row) x = -__builtin_inff();
          cv[u] = x;
        }
        float tmax = fmaxf(fmaxf(cv[0], cv[1]), fmaxf(cv[2], cv[3]));
#pragma unroll
        for (int mk = 1; mk < 16; mk <<= 1) tmax = fmaxf(tmax, __shfl_xor(tmax, mk, 32));
        float mnew = fmaxf(m[j], tmax);
        float p = __expf(cv[0] - mnew) + __expf(cv[1] - mnew) +
                  __expf(cv[2] - mnew) + __expf(cv[3] - mnew);
#pragma unroll
        for (int mk = 1; mk < 16; mk <<= 1) p += __shfl_xor(p, mk, 32);
        r[j] = r[j] * __expf(m[j] - mnew) + p;
        m[j] = mnew;
      }
    }
    asm volatile("s_wait_dscnt 0" ::: "memory");
    __syncthreads();                       // done reading buf before reuse
  }
  float inv[8];
#pragma unroll
  for (int j = 0; j < 8; ++j) inv[j] = 1.f / r[j];

  // ================= sweep B: attn write + fused P@V =================
  // prologue: stage 0 K + V copies (4 async instrs per thread)
  {
    const _Float16* gk = kbase + (size_t)srow * DM + sseg * 8;
    async_copy16(gk,      &kst[0][srow * 64 + sseg * 8]);
    async_copy16(gk + 32, &kst[0][srow * 64 + sseg * 8 + 32]);
    const _Float16* gv = vhead + (size_t)srow * SEQ + sseg * 8;
    async_copy16(gv,      &vst[0][srow * 64 + sseg * 8]);
    async_copy16(gv + 32, &vst[0][srow * 64 + sseg * 8 + 32]);
  }
  v8f oacc[4] = {};
  for (int sg = 0; sg < NSTAGE; ++sg) {
    int buf = sg & 1;
    if (sg + 1 < NSTAGE) {
      const _Float16* gk = kbase + (size_t)((sg + 1) * 64 + srow) * DM + sseg * 8;
      async_copy16(gk,      &kst[buf ^ 1][srow * 64 + sseg * 8]);
      async_copy16(gk + 32, &kst[buf ^ 1][srow * 64 + sseg * 8 + 32]);
      const _Float16* gv = vhead + (size_t)srow * SEQ + (sg + 1) * 64 + sseg * 8;
      async_copy16(gv,      &vst[buf ^ 1][srow * 64 + sseg * 8]);
      async_copy16(gv + 32, &vst[buf ^ 1][srow * 64 + sseg * 8 + 32]);
      asm volatile("s_wait_asynccnt 4" ::: "memory");
    } else {
      asm volatile("s_wait_asynccnt 0" ::: "memory");
    }
    __syncthreads();
    int sbase = sg * 64;
    // two 32-wide pairs per stage
#pragma unroll
    for (int p2 = 0; p2 < 2; ++p2) {
      int spair = sbase + p2 * 32;
      if (spair >= nt * 16) continue;      // wave-local causal skip (no barriers inside)
      for (int tt = 0; tt < 2; ++tt) {
        int st = (sg * 4) + p2 * 2 + tt;
        if (st < nt) {
          const _Float16* cp = &kst[buf][((p2 * 2 + tt) * 16) * 64];
          v16h k0 = loadB32x16(cp, 64, lane);
          v16h k1 = loadB32x16(cp + 32, 64, lane);
          v8f c = {};
          c = wmma_f16(qa0, k0, c);
          c = wmma_f16(qa1, k1, c);
          int sc = st * 16;
#pragma unroll
          for (int j = 0; j < 8; ++j) {
            float cv = c[j];
            if (sc + n > t0 + half8 + j) cv = -__builtin_inff();
            float p = __expf(cv - m[j]) * inv[j];
            arow[(size_t)(t0 + half8 + j) * SEQ + sc + n] = p;  // coalesced rows
            plds[wv][(half8 + j) * 40 + tt * 16 + n] = (_Float16)p;
          }
        } else {
#pragma unroll
          for (int j = 0; j < 8; ++j)
            plds[wv][(half8 + j) * 40 + tt * 16 + n] = (_Float16)0.f;
        }
      }
      // wave-local LDS transpose fence (C-layout -> A-layout)
      asm volatile("s_wait_dscnt 0" ::: "memory");
      v16h pa = loadA16x32(&plds[wv][0], 40, lane);
#pragma unroll
      for (int dt = 0; dt < 4; ++dt) {
        v16h vb = loadB32x16(&vst[buf][(dt * 16) * 64 + p2 * 32], 64, lane);
        oacc[dt] = wmma_f16(pa, vb, oacc[dt]);
      }
    }
    asm volatile("s_wait_dscnt 0" ::: "memory");
    __syncthreads();
  }

  // ---- write per-head context (already normalized) as f16 [B,T,D] ----
  _Float16* crow = ctx + ((size_t)(b * SEQ) + t0) * DM + h * HDIM;
#pragma unroll
  for (int dt = 0; dt < 4; ++dt)
#pragma unroll
    for (int j = 0; j < 8; ++j)
      crow[(size_t)(half8 + j) * DM + dt * 16 + n] = (_Float16)oacc[dt][j];

  // ---- zero-fill the unvisited upper-triangular attn region ----
  int sfill = nt * 16;
  for (int j = 0; j < 16; ++j) {
    float* rp = arow + (size_t)(t0 + j) * SEQ;
    for (int s = sfill + lane * 4; s < SEQ; s += 128) {
      float4 z = make_float4(0.f, 0.f, 0.f, 0.f);
      *(float4*)(rp + s) = z;
    }
  }
}

// ---------------- host-side launcher ----------------
extern "C" void kernel_launch(void* const* d_in, const int* in_sizes, int n_in,
                              void* d_out, int out_size, void* d_ws, size_t ws_size,
                              hipStream_t stream) {
  (void)in_sizes; (void)n_in; (void)out_size; (void)ws_size;
  const float* query = (const float*)d_in[0];
  const float* key_  = (const float*)d_in[1];
  const float* value = (const float*)d_in[2];
  const float* Wq = (const float*)d_in[3];  const float* bq = (const float*)d_in[4];
  const float* Wk = (const float*)d_in[5];  const float* bk = (const float*)d_in[6];
  const float* Wv = (const float*)d_in[7];  const float* bv = (const float*)d_in[8];
  const float* Wo = (const float*)d_in[9];  const float* bo = (const float*)d_in[10];
  const float* ln_g = (const float*)d_in[11];
  const float* ln_b = (const float*)d_in[12];

  float* out_main = (float*)d_out;                              // [B,T,D]
  float* out_attn = (float*)d_out + (size_t)ROWS * DM;          // [B,H,T,S]

  _Float16* ws = (_Float16*)d_ws;
  const size_t ACT = (size_t)ROWS * DM;   // 4,194,304 halves
  const size_t WSZ = (size_t)DM * DM;     // 1,048,576 halves
  _Float16* xq   = ws;
  _Float16* xk   = ws + ACT;
  _Float16* xv   = ws + 2 * ACT;
  _Float16* qbuf = ws + 3 * ACT;
  _Float16* kbuf = ws + 4 * ACT;
  _Float16* vtb  = ws + 5 * ACT;
  _Float16* ctx  = ws + 6 * ACT;
  _Float16* wq16 = ws + 7 * ACT;
  _Float16* wk16 = wq16 + WSZ;
  _Float16* wv16 = wk16 + WSZ;
  _Float16* wo16 = wv16 + WSZ;

  ln_kernel<<<ROWS, 256, 0, stream>>>(query, ln_g, ln_b, xq);
  ln_kernel<<<ROWS, 256, 0, stream>>>(key_,  ln_g, ln_b, xk);
  ln_kernel<<<ROWS, 256, 0, stream>>>(value, ln_g, ln_b, xv);

  int nv4 = (DM * DM) / 4;
  cvt_f32_f16<<<nv4 / 256, 256, 0, stream>>>(Wq, wq16, nv4);
  cvt_f32_f16<<<nv4 / 256, 256, 0, stream>>>(Wk, wk16, nv4);
  cvt_f32_f16<<<nv4 / 256, 256, 0, stream>>>(Wv, wv16, nv4);
  cvt_f32_f16<<<nv4 / 256, 256, 0, stream>>>(Wo, wo16, nv4);

  gemm_nt<<<512, 256, 0, stream>>>(xq, wq16, bq, (void*)qbuf, QSCALE, 0);
  gemm_nt<<<512, 256, 0, stream>>>(xk, wk16, bk, (void*)kbuf, 1.0f,   0);
  gemm_nt<<<512, 256, 0, stream>>>(xv, wv16, bv, (void*)vtb,  1.0f,   1);

  attn_kernel<<<dim3(SEQ / 128, BATCH * NHEAD), 256, 0, stream>>>(
      qbuf, kbuf, vtb, ctx, out_attn);

  gemm_nt<<<512, 256, 0, stream>>>(ctx, wo16, bo, (void*)out_main, 1.0f, 2);
}
